// SparseNeighborhoodAttentionBlock_81664508166253
// MI455X (gfx1250) — matrix operand order repaired and checked by reading, not compiled, not run on hardware
//
#include <hip/hip_runtime.h>
#include <hip/hip_bf16.h>
#include <math.h>

// Problem constants (from reference)
#define NQ     2048
#define DMODEL 256
#define NHEAD  8
#define HDIM   32
#define HALF   16
#define LVLS   4
#define KKEYS  164
#define KPAD   176
#define HMAXC  64
#define ROWH   264          // padded LDS row length in bf16 elements (528B: 16B-aligned, bank-staggered)
#define LN_EPS 1e-5f

typedef __attribute__((ext_vector_type(16))) __bf16 v16bf;
typedef __attribute__((ext_vector_type(8)))  float  v8f;

union FragBF { uint4 u[2]; v16bf v; };

__device__ __forceinline__ unsigned short f2bf(float f) {
  unsigned int u = __float_as_uint(f);
  u += 0x7fffu + ((u >> 16) & 1u);          // round-to-nearest-even
  return (unsigned short)(u >> 16);
}
__device__ __forceinline__ float bf2f(unsigned short h) {
  return __uint_as_float(((unsigned int)h) << 16);
}
__device__ __forceinline__ unsigned int pack2(float a, float b) {
  return (unsigned int)f2bf(a) | ((unsigned int)f2bf(b) << 16);
}

// ---------------- Prep: Wk = Wkv[:, :256] -> bf16, transposed to [n][k] ----------------
__global__ __launch_bounds__(256)
void prep_wk_bf16(const float* __restrict__ Wkv, unsigned short* __restrict__ WkT) {
  int idx = blockIdx.x * 256 + threadIdx.x;        // 256*256 elements
  int n = idx >> 8;
  int k = idx & 255;
  WkT[n * 256 + k] = f2bf(Wkv[k * 512 + n]);
}

// Dynamic LDS layout (bytes)
#define SM_VAL   0
#define SM_KMAT  (SM_VAL  + KPAD * ROWH * 2)
#define SM_KEYP  (SM_KMAT + KPAD * ROWH * 2)
#define SM_QN    (SM_KEYP + KPAD * 4 * 4)
#define SM_QRAW  (SM_QN   + DMODEL * 4)
#define SM_QROPE (SM_QRAW + DMODEL * 4)
#define SM_SCORE (SM_QROPE + DMODEL * 4)
#define SM_WTS   (SM_SCORE + NHEAD * KPAD * 4)
#define SM_TBUF  (SM_WTS   + NHEAD * KPAD * 4)
#define SM_XBUF  (SM_TBUF  + NHEAD * DMODEL * 4)
#define SM_RED   (SM_XBUF  + DMODEL * 4)
#define SM_TOTAL (SM_RED   + 64)

__device__ __forceinline__ float block_sum256(float v, float* red, int tid) {
  #pragma unroll
  for (int off = 16; off; off >>= 1) v += __shfl_xor(v, off, 32);
  if ((tid & 31) == 0) red[tid >> 5] = v;
  __syncthreads();
  float tot = 0.f;
  #pragma unroll
  for (int i = 0; i < 8; i++) tot += red[i];
  __syncthreads();
  return tot;
}

// ---------------- Main fused kernel: one workgroup (8 wave32) per query ----------------
__global__ __launch_bounds__(256)
void sna_main(const float* __restrict__ query,
              const float* __restrict__ qpos,        // (N,4): b,i,j,lev
              const float* __restrict__ fmap,        // (B,64,64,L,D)
              const int*   __restrict__ shapes,      // (4,2)
              const float* __restrict__ gamma,
              const float* __restrict__ beta,
              const float* __restrict__ Wq,          // (D,D) row-major
              const float* __restrict__ Wkv,         // (D,2D) row-major
              const float* __restrict__ Wout,        // (D,D)
              const float* __restrict__ freqs,       // (NH,3,16)
              const unsigned short* __restrict__ WkT,// (256,256) bf16, [n][k]
              float* __restrict__ out) {
  extern __shared__ char smem[];
  unsigned short* valmat = (unsigned short*)(smem + SM_VAL);
  unsigned short* kmat   = (unsigned short*)(smem + SM_KMAT);
  float* keyp   = (float*)(smem + SM_KEYP);   // (KPAD,4): i_pos, j_pos, lev, valid
  float* qn     = (float*)(smem + SM_QN);
  float* qraw   = (float*)(smem + SM_QRAW);
  float* qrope  = (float*)(smem + SM_QROPE);
  float* scores = (float*)(smem + SM_SCORE);  // (NH,KPAD)
  float* wts    = (float*)(smem + SM_WTS);    // (NH,KPAD)
  float* tbuf   = (float*)(smem + SM_TBUF);   // (NH,D)
  float* xbuf   = (float*)(smem + SM_XBUF);
  float* red    = (float*)(smem + SM_RED);

  const int n    = blockIdx.x;
  const int tid  = threadIdx.x;
  const int lane = tid & 31;
  const int wave = tid >> 5;

  const float qpi = qpos[n * 4 + 1];
  const float qpj = qpos[n * 4 + 2];
  const float qpl = qpos[n * 4 + 3];
  const int   bb  = (int)qpos[n * 4 + 0];

  int sh_i[4], sh_j[4];
  #pragma unroll
  for (int m = 0; m < 4; m++) { sh_i[m] = shapes[m * 2]; sh_j[m] = shapes[m * 2 + 1]; }
  float sci[4], scj[4];
  #pragma unroll
  for (int m = 0; m < 4; m++) {
    sci[m] = (float)sh_i[m] / (float)sh_i[3];
    scj[m] = (float)sh_j[m] / (float)sh_j[3];
  }

  // ---- Phase 0: per-key metadata + masked gather of value rows into LDS (bf16) ----
  // wave handles keys wave, wave+8, ... ; 32 lanes cover the 256-wide row (8 floats each)
  for (int key = wave; key < KPAD; key += 8) {
    uint4 z; z.x = z.y = z.z = z.w = 0u;
    if (key < KKEYS) {
      int m, base, s;
      if (key < 9)       { m = 0; base = 0;  s = 3; }
      else if (key < 34) { m = 1; base = 9;  s = 5; }
      else if (key < 83) { m = 2; base = 34; s = 7; }
      else               { m = 3; base = 83; s = 9; }
      int d  = key - base;
      int di = d / s - (s - 1) / 2;
      int dj = d % s - (s - 1) / 2;
      int ci = (int)floorf(qpi * sci[m]);
      int cj = (int)floorf(qpj * scj[m]);
      int ii = ci + di, jj = cj + dj;
      bool valid = (ii >= 0) && (ii < sh_i[m]) && (jj >= 0) && (jj < sh_j[m]);
      int ic = min(max(ii, 0), HMAXC - 1);
      int jc = min(max(jj, 0), HMAXC - 1);
      if (lane == 0) {
        keyp[key * 4 + 0] = (float)ii / sci[m];
        keyp[key * 4 + 1] = (float)jj / scj[m];
        keyp[key * 4 + 2] = (float)m;
        keyp[key * 4 + 3] = valid ? 1.f : 0.f;
      }
      const float4* src = (const float4*)(fmap +
          ((((size_t)bb * HMAXC + ic) * HMAXC + jc) * LVLS + m) * DMODEL);
      float vf = valid ? 1.f : 0.f;
      float4 a = src[lane * 2];
      float4 b = src[lane * 2 + 1];
      uint4 p;
      p.x = pack2(a.x * vf, a.y * vf);
      p.y = pack2(a.z * vf, a.w * vf);
      p.z = pack2(b.x * vf, b.y * vf);
      p.w = pack2(b.z * vf, b.w * vf);
      *(uint4*)(valmat + key * ROWH + lane * 8) = p;
    } else {
      *(uint4*)(valmat + key * ROWH + lane * 8) = z;   // zero pad rows 164..175
    }
  }
  __syncthreads();

  // ---- Phase 1: LayerNorm(query) -> qn ----
  float qv  = query[n * DMODEL + tid];
  float mu  = block_sum256(qv, red, tid) * (1.f / DMODEL);
  float dif = qv - mu;
  float var = block_sum256(dif * dif, red, tid) * (1.f / DMODEL);
  float qnv = dif * rsqrtf(var + LN_EPS) * gamma[tid] + beta[tid];
  qn[tid] = qnv;
  __syncthreads();

  // ---- Phase 2: q = qn @ Wq (GEMV, 131 KFLOP) ----
  {
    float acc = 0.f;
    for (int d = 0; d < DMODEL; d++) acc += qn[d] * Wq[d * DMODEL + tid];
    qraw[tid] = acc;
  }
  __syncthreads();

  // ---- Phase 3: RoPE(q) with query position (i, j, lev) ----
  {
    int h = tid >> 5;
    int f = (tid & 31) >> 1;
    const float* fr = freqs + h * 48;
    float ang = qpi * fr[f] + qpj * fr[16 + f] + qpl * fr[32 + f];
    float c = cosf(ang), s = sinf(ang);
    float x1 = qraw[tid & ~1], x2 = qraw[tid | 1];
    qrope[tid] = (tid & 1) ? (x1 * s + x2 * c) : (x1 * c - x2 * s);
  }
  __syncthreads();

  // ---- Phase 4: k = value @ Wk via WMMA bf16 (M=176, N=256, K=256) ----
  // 11x16 = 176 output tiles processed as 88 adjacent-N pairs sharing the A fragment.
  // A (16-bit 16x32): lanes 0-15 K-chunks {0..7,16..23}, lanes 16-31 {8..15,24..31}.
  // B (16-bit 32x16): lane holds col n=lane%16, same K-chunks as A: {0..7,16..23} / {8..15,24..31}.
  {
    const int hi   = lane >> 4;      // half-wave id
    const int lrow = lane & 15;
    for (int p = wave; p < 88; p += 8) {
      int t0 = 2 * p;
      int mt = t0 >> 4;
      int nt0 = t0 & 15;
      const unsigned short* arow = valmat + (mt * 16 + lrow) * ROWH;
      const unsigned short* brow0 = WkT + (nt0 * 16 + lrow) * 256;
      const unsigned short* brow1 = brow0 + 16 * 256;
      v8f acc0 = {0.f, 0.f, 0.f, 0.f, 0.f, 0.f, 0.f, 0.f};
      v8f acc1 = {0.f, 0.f, 0.f, 0.f, 0.f, 0.f, 0.f, 0.f};
      #pragma unroll
      for (int ks = 0; ks < 8; ks++) {
        int kb = ks * 32;
        FragBF a, b0, b1;
        a.u[0]  = *(const uint4*)(arow  + kb + hi * 8);
        a.u[1]  = *(const uint4*)(arow  + kb + 16 + hi * 8);
        b0.u[0] = *(const uint4*)(brow0 + kb + hi * 8);
        b0.u[1] = *(const uint4*)(brow0 + kb + 16 + hi * 8);
        b1.u[0] = *(const uint4*)(brow1 + kb + hi * 8);
        b1.u[1] = *(const uint4*)(brow1 + kb + 16 + hi * 8);
        acc0 = __builtin_amdgcn_wmma_f32_16x16x32_bf16(false, a.v, false, b0.v,
                                                       (short)0, acc0, false, false);
        acc1 = __builtin_amdgcn_wmma_f32_16x16x32_bf16(false, a.v, false, b1.v,
                                                       (short)0, acc1, false, false);
      }
      // C layout: col = lane%16; VGPR r -> row r (lanes 0-15) / 8+r (lanes 16-31)
      int rbase = mt * 16 + hi * 8;
      int col0 = nt0 * 16 + lrow;
      #pragma unroll
      for (int r = 0; r < 8; r++) {
        kmat[(rbase + r) * ROWH + col0]      = f2bf(acc0[r]);
        kmat[(rbase + r) * ROWH + col0 + 16] = f2bf(acc1[r]);
      }
    }
  }
  __syncthreads();

  // ---- Phase 5: scores = RoPE(k) . qrope, masked ----
  for (int t = tid; t < KPAD * NHEAD; t += 256) {
    int key = t >> 3;
    int h   = t & 7;
    float sc;
    if (key >= KKEYS || keyp[key * 4 + 3] == 0.f) {
      sc = -__builtin_inff();
    } else {
      float pi = keyp[key * 4 + 0], pj = keyp[key * 4 + 1], pl = keyp[key * 4 + 2];
      const unsigned short* krow = kmat + key * ROWH + h * HDIM;
      const float* fr = freqs + h * 48;
      float s_ = 0.f;
      #pragma unroll
      for (int f = 0; f < HALF; f++) {
        float ang = pi * fr[f] + pj * fr[16 + f] + pl * fr[32 + f];
        float c = cosf(ang), s2 = sinf(ang);
        float k1 = bf2f(krow[2 * f]), k2 = bf2f(krow[2 * f + 1]);
        float q1 = qrope[h * HDIM + 2 * f], q2 = qrope[h * HDIM + 2 * f + 1];
        s_ += (k1 * c - k2 * s2) * q1 + (k1 * s2 + k2 * c) * q2;
      }
      sc = s_;
    }
    scores[h * KPAD + key] = sc;
  }
  __syncthreads();

  // ---- Phase 6: per-head softmax over keys (one wave per head) ----
  {
    int h = wave;
    float mx = -__builtin_inff();
    for (int k0 = lane; k0 < KPAD; k0 += 32) mx = fmaxf(mx, scores[h * KPAD + k0]);
    #pragma unroll
    for (int off = 16; off; off >>= 1) mx = fmaxf(mx, __shfl_xor(mx, off, 32));
    float sum = 0.f;
    for (int k0 = lane; k0 < KPAD; k0 += 32) {
      float e = __expf(scores[h * KPAD + k0] - mx);
      wts[h * KPAD + k0] = e;
      sum += e;
    }
    #pragma unroll
    for (int off = 16; off; off >>= 1) sum += __shfl_xor(sum, off, 32);
    float inv = 1.f / sum;
    for (int k0 = lane; k0 < KPAD; k0 += 32) wts[h * KPAD + k0] *= inv;
  }
  __syncthreads();

  // ---- Phase 7: t[h,:] = w[h,:] @ value  (reordered weighted sum, before V-proj) ----
  for (int t = tid; t < NHEAD * DMODEL; t += 256) {
    int h = t >> 8;
    int d = t & 255;
    float acc = 0.f;
    for (int k0 = 0; k0 < KKEYS; k0++)
      acc += wts[h * KPAD + k0] * bf2f(valmat[k0 * ROWH + d]);
    tbuf[h * DMODEL + d] = acc;
  }
  __syncthreads();

  // ---- Phase 8: x[j] = t[h,:] @ Wv[:, j]  (Wv = Wkv[:, 256:512]) ----
  {
    int h = tid >> 5;
    float acc = 0.f;
    const float* th = tbuf + h * DMODEL;
    for (int d = 0; d < DMODEL; d++) acc += th[d] * Wkv[d * 512 + 256 + tid];
    xbuf[tid] = acc;
  }
  __syncthreads();

  // ---- Phase 9: out = x @ Wout + residual ----
  {
    float acc = 0.f;
    for (int d = 0; d < DMODEL; d++) acc += xbuf[d] * Wout[d * DMODEL + tid];
    out[n * DMODEL + tid] = acc + query[n * DMODEL + tid];
  }
}

extern "C" void kernel_launch(void* const* d_in, const int* in_sizes, int n_in,
                              void* d_out, int out_size, void* d_ws, size_t ws_size,
                              hipStream_t stream) {
  const float* query  = (const float*)d_in[0];
  const float* qpos   = (const float*)d_in[1];
  // d_in[2] = query_batch_offsets (unused; batch derivable from qpos col 0)
  const float* fmap   = (const float*)d_in[3];
  const int*   shapes = (const int*)d_in[4];
  const float* gamma  = (const float*)d_in[5];
  const float* beta   = (const float*)d_in[6];
  const float* Wq     = (const float*)d_in[7];
  const float* Wkv    = (const float*)d_in[8];
  const float* Wout   = (const float*)d_in[9];
  const float* freqs  = (const float*)d_in[10];
  float* out = (float*)d_out;

  unsigned short* WkT = (unsigned short*)d_ws;   // 256*256 bf16 = 128 KB

  prep_wk_bf16<<<dim3((256 * 256) / 256), dim3(256), 0, stream>>>(Wkv, WkT);
  sna_main<<<dim3(NQ), dim3(256), SM_TOTAL, stream>>>(
      query, qpos, fmap, shapes, gamma, beta, Wq, Wkv, Wout, freqs, WkT, out);
}
